// MultiHeadAttention_48902497632900
// MI455X (gfx1250) — compile-verified, matrix operand order checked by reading, not run on hardware
//
#include <hip/hip_runtime.h>

// ---------------- problem constants ----------------
#define D_MODEL   1024
#define NUM_HEADS 16
#define D_K       64
#define BATCH     2
#define SEQ       2048
#define NROWS     (BATCH * SEQ)          // 4096 flattened (b,t) rows

typedef __bf16 bf16;
typedef __bf16 v16bf __attribute__((ext_vector_type(16)));
typedef __bf16 v8bf  __attribute__((ext_vector_type(8)));
typedef float  v8f   __attribute__((ext_vector_type(8)));
typedef unsigned int v4u __attribute__((ext_vector_type(4)));
typedef int v4i __attribute__((ext_vector_type(4)));
typedef int v8i __attribute__((ext_vector_type(8)));

// ---------------- small helpers ----------------
__device__ __forceinline__ v8f vzero8() {
    v8f z;
#pragma unroll
    for (int i = 0; i < 8; ++i) z[i] = 0.0f;
    return z;
}

// Load a 16-element bf16 fragment as two contiguous 16-byte chunks.
// Per the CDNA5 16-bit A/B fragment layout, each lane holds two contiguous
// 8-element K-runs (K = half*8..+7 and 16+half*8..+7), so two b128 loads.
__device__ __forceinline__ v16bf load_frag(const bf16* p0, const bf16* p1) {
    v8bf lo = *reinterpret_cast<const v8bf*>(p0);
    v8bf hi = *reinterpret_cast<const v8bf*>(p1);
    v16bf r;
#pragma unroll
    for (int i = 0; i < 8; ++i) { r[i] = lo[i]; r[i + 8] = hi[i]; }
    return r;
}

__device__ __forceinline__ v8f wmma_bf16(v16bf a, v16bf b, v8f c) {
    // D(16x16,f32) = A(16x32,bf16) * B(32x16,bf16) + C
    return __builtin_amdgcn_wmma_f32_16x16x32_bf16(
        /*neg_a=*/false, a, /*neg_b=*/false, b,
        /*c_mod=*/(short)0, c, /*reuse_a=*/false, /*reuse_b=*/false);
}

// Tensor Data Mover: async DMA of a 2D bf16 tile (tile_d1 rows x tile_d0 cols,
// row stride stride_d0 elements) from global memory into LDS at lds_addr.
// Descriptor packing per CDNA5 ISA.
// Tracked by TENSORcnt; pair with __builtin_amdgcn_s_wait_tensorcnt(0).
// This toolchain exposes the 6-arg builtin:
//   (uint32x4 g0, int32x8 g1, int32x4 g2, int32x4 g3, int32x8 extra, i32 cpol)
__device__ __forceinline__ void tdm_load_2d_bf16(unsigned lds_addr, const void* gaddr,
                                                 unsigned tile_d0, unsigned tile_d1,
                                                 unsigned tensor_d0, unsigned tensor_d1,
                                                 unsigned stride_d0) {
    const unsigned long long ga = (unsigned long long)gaddr;
    v4u g0;
    g0[0] = 1u;                                     // count=1, user descriptor
    g0[1] = lds_addr;                               // LDS byte address
    g0[2] = (unsigned)(ga & 0xffffffffu);           // global_addr[31:0]
    g0[3] = (unsigned)((ga >> 32) & 0x01ffffffu)    // global_addr[56:32]
            | (2u << 30);                           // type=2 ("image")
    v8i g1;
    g1[0] = (int)(1u << 16);                        // data_size=1 (2 bytes/elem)
    g1[1] = (int)(tensor_d0 << 16);                 // tensor_dim0[15:0] @ bits 63:48
    g1[2] = (int)((tensor_d0 >> 16) | (tensor_d1 << 16));   // dim0 hi | dim1 lo
    g1[3] = (int)((tensor_d1 >> 16) | (tile_d0 << 16));     // dim1 hi | tile_dim0
    g1[4] = (int)(tile_d1 & 0xffffu);               // tile_dim1 (tile_dim2 = 0)
    g1[5] = (int)stride_d0;                         // tensor_dim0_stride[31:0]
    g1[6] = 0;                                      // stride hi, dim1_stride lo
    g1[7] = 0;
    const v4i z4 = {0, 0, 0, 0};                    // groups 2/3 unused (2D)
    const v8i z8 = {0, 0, 0, 0, 0, 0, 0, 0};
    __builtin_amdgcn_tensor_load_to_lds(g0, g1, z4, z4, z8, 0);
}

// ---------------- kernel 1: fp32 -> bf16 convert ----------------
__global__ void cvt_f32_bf16_kernel(const float* __restrict__ src,
                                    bf16* __restrict__ dst, int n) {
    int i = blockIdx.x * blockDim.x + threadIdx.x;
    int stride = gridDim.x * blockDim.x;
    for (; i < n; i += stride) dst[i] = (bf16)src[i];
}

// ---------------- kernel 2: per-head Q/K/V projection ----------------
// grid = (NROWS/128, NUM_HEADS, 3), block = 256 (8 waves).
__global__ void proj_kernel(const bf16* __restrict__ Qx, const bf16* __restrict__ Kx,
                            const bf16* __restrict__ Vx,
                            const bf16* __restrict__ Wq, const bf16* __restrict__ Wk,
                            const bf16* __restrict__ Wv,
                            const float* __restrict__ bq, const float* __restrict__ bk,
                            const float* __restrict__ bv,
                            bf16* __restrict__ Qh, bf16* __restrict__ Kh,
                            bf16* __restrict__ Vh) {
    const int which = blockIdx.z;
    const bf16*  X    = (which == 0) ? Qx : (which == 1) ? Kx : Vx;
    const bf16*  W    = (which == 0) ? Wq : (which == 1) ? Wk : Wv;
    const float* bias = (which == 0) ? bq : (which == 1) ? bk : bv;
    bf16*        Out  = (which == 0) ? Qh : (which == 1) ? Kh : Vh;

    const int h = blockIdx.y;
    W    += (size_t)h * D_MODEL * D_K;
    bias += h * D_K;

    const int tid  = threadIdx.x;
    const int wave = tid >> 5;
    const int lane = tid & 31;
    const int half = lane >> 4;
    const int nl   = lane & 15;
    const int r0   = blockIdx.x * 128 + wave * 16;

    // W tile staged transposed: Wt[n][k], n in [0,64), k in [0,32), row pitch 40
    __shared__ __align__(16) bf16 Wt[64 * 40];

    v8f acc[4];
#pragma unroll
    for (int j = 0; j < 4; ++j) acc[j] = vzero8();

    const bf16* arow = X + (size_t)(r0 + nl) * D_MODEL;

    for (int kk = 0; kk < D_MODEL; kk += 32) {
        // cooperative transpose-stage of W[kk..kk+31][0..63]
        {
            const int k = tid >> 3;            // 0..31
            const int n = (tid & 7) * 8;       // 0,8,..,56
            v8bf w = *reinterpret_cast<const v8bf*>(W + (size_t)(kk + k) * D_K + n);
#pragma unroll
            for (int i = 0; i < 8; ++i) Wt[(n + i) * 40 + k] = w[i];
        }
        __syncthreads();

        v16bf a = load_frag(arow + kk + half * 8, arow + kk + 16 + half * 8);
#pragma unroll
        for (int j = 0; j < 4; ++j) {
            const bf16* bp = Wt + (j * 16 + nl) * 40 + half * 8;
            v16bf b = load_frag(bp, bp + 16);
            acc[j] = wmma_bf16(a, b, acc[j]);
        }
        __syncthreads();
    }

    // bias add + store bf16 into [B,H,T,D_K]
#pragma unroll
    for (int j = 0; j < 4; ++j) {
        const float bb = bias[j * 16 + nl];
#pragma unroll
        for (int r = 0; r < 8; ++r) {
            const int g = r0 + r + 8 * half;       // flattened (b,t)
            const int b = g >> 11;                 // /SEQ
            const int t = g & (SEQ - 1);
            Out[(((size_t)b * NUM_HEADS + h) * SEQ + t) * D_K + j * 16 + nl] =
                (bf16)(acc[j][r] + bb);
        }
    }
}

// ---------------- kernel 3: flash attention per (b,h) ----------------
// grid = B*H*(SEQ/128) = 512, block = 256 (8 waves, 16 query rows each).
// K chunks are DMA'd into LDS by the Tensor Data Mover (wave 0) while all
// waves cooperatively transpose-stage the V chunk; TENSORcnt + workgroup
// barrier joins the two paths.
__global__ void attn_kernel(const bf16* __restrict__ Qh, const bf16* __restrict__ Kh,
                            const bf16* __restrict__ Vh, bf16* __restrict__ Ctx) {
    const int bh = blockIdx.x >> 4;        // b*H + h  (0..31)
    const int qg = blockIdx.x & 15;
    const int tid  = threadIdx.x;
    const int wave = tid >> 5;
    const int lane = tid & 31;
    const int half = lane >> 4;
    const int nl   = lane & 15;
    const int q0   = qg * 128 + wave * 16;

    const bf16* Qb = Qh + (size_t)bh * SEQ * D_K;
    const bf16* Kb = Kh + (size_t)bh * SEQ * D_K;
    const bf16* Vb = Vh + (size_t)bh * SEQ * D_K;

    __shared__ __align__(16) bf16 Kch[32 * 64];      // K chunk, row-major [s][d] (TDM dest)
    __shared__ __align__(16) bf16 Vt[64 * 40];       // V chunk transposed [d][s]
    __shared__ __align__(16) bf16 Pscr[8 * 16 * 32]; // per-wave P relayout scratch
    bf16* Pw = Pscr + wave * 16 * 32;
    const unsigned kch_lds = (unsigned)(unsigned long long)(void*)Kch;

    // Q A-fragments (16x32 each), loaded once
    v16bf qa[2];
#pragma unroll
    for (int kt = 0; kt < 2; ++kt) {
        const bf16* p = Qb + (size_t)(q0 + nl) * D_K + kt * 32 + half * 8;
        qa[kt] = load_frag(p, p + 16);
    }

    v8f c[4];
#pragma unroll
    for (int v = 0; v < 4; ++v) c[v] = vzero8();
    float rmax[8], rsum[8];
#pragma unroll
    for (int r = 0; r < 8; ++r) { rmax[r] = -3.0e38f; rsum[r] = 0.0f; }

    const float scale = 0.125f;  // 1/sqrt(64)

    for (int s0 = 0; s0 < SEQ; s0 += 32) {
        // TDM: async-load K[s0..s0+31][0..63] into Kch (issued once, by wave 0)
        if (wave == 0) {
            tdm_load_2d_bf16(kch_lds, Kb + (size_t)s0 * D_K,
                             /*tile_d0=*/D_K, /*tile_d1=*/32,
                             /*tensor_d0=*/D_K, /*tensor_d1=*/SEQ,
                             /*stride_d0=*/D_K);
        }
        // Meanwhile: all 256 threads transpose-stage the V chunk
        {
            const int r    = tid >> 3;        // 0..31 (key within chunk)
            const int ccol = (tid & 7) * 8;   // 0..56 (d_k)
            v8bf v = *reinterpret_cast<const v8bf*>(Vb + (size_t)(s0 + r) * D_K + ccol);
#pragma unroll
            for (int i = 0; i < 8; ++i) Vt[(ccol + i) * 40 + r] = v[i];
        }
        if (s0 + 32 < SEQ)
            __builtin_prefetch(Vb + (size_t)(s0 + 32) * D_K + lane * 64, 0, 1);
        if (wave == 0)
            __builtin_amdgcn_s_wait_tensorcnt(0);  // K chunk resident in LDS
        __syncthreads();

        // S = Q * K^T for this 16x32 tile (two 16x16 D tiles)
        v8f s[2];
#pragma unroll
        for (int j = 0; j < 2; ++j) {
            v8f a = vzero8();
#pragma unroll
            for (int kt = 0; kt < 2; ++kt) {
                const bf16* bp = Kch + (j * 16 + nl) * 64 + kt * 32 + half * 8;
                v16bf b = load_frag(bp, bp + 16);
                a = wmma_bf16(qa[kt], b, a);
            }
            s[j] = a;
        }

        // online softmax (rows live across 16 lanes of each half-wave)
#pragma unroll
        for (int r = 0; r < 8; ++r) {
            float x0 = s[0][r] * scale;
            float x1 = s[1][r] * scale;
            float t = fmaxf(x0, x1);
            t = fmaxf(t, __shfl_xor(t, 8, 16));
            t = fmaxf(t, __shfl_xor(t, 4, 16));
            t = fmaxf(t, __shfl_xor(t, 2, 16));
            t = fmaxf(t, __shfl_xor(t, 1, 16));
            const float mnew  = fmaxf(rmax[r], t);
            const float alpha = __expf(rmax[r] - mnew);
            rmax[r] = mnew;
            const float p0 = __expf(x0 - mnew);
            const float p1 = __expf(x1 - mnew);
            float ss = p0 + p1;
            ss += __shfl_xor(ss, 8, 16);
            ss += __shfl_xor(ss, 4, 16);
            ss += __shfl_xor(ss, 2, 16);
            ss += __shfl_xor(ss, 1, 16);
            rsum[r] = rsum[r] * alpha + ss;
#pragma unroll
            for (int v = 0; v < 4; ++v) c[v][r] *= alpha;
            // spill P to wave-private LDS in [m][k] order for A-frag reload
            const int m = r + 8 * half;
            Pw[m * 32 + nl]      = (bf16)p0;
            Pw[m * 32 + 16 + nl] = (bf16)p1;
        }

        // ctx += P(16x32) * V(32x64); LDS ops from same wave are in-order
        {
            const bf16* pp = Pw + nl * 32 + half * 8;
            v16bf pa = load_frag(pp, pp + 16);
#pragma unroll
            for (int v = 0; v < 4; ++v) {
                const bf16* vp = Vt + (v * 16 + nl) * 40 + half * 8;
                v16bf vb = load_frag(vp, vp + 16);
                c[v] = wmma_bf16(pa, vb, c[v]);
            }
        }
        __syncthreads();
    }

    // normalize and store ctx (bf16, [B,H,T,D_K] contiguous)
#pragma unroll
    for (int r = 0; r < 8; ++r) {
        const float inv = 1.0f / rsum[r];
        const int trow = q0 + r + 8 * half;
#pragma unroll
        for (int v = 0; v < 4; ++v)
            Ctx[((size_t)bh * SEQ + trow) * D_K + v * 16 + nl] =
                (bf16)(c[v][r] * inv);
    }
}

// ---------------- kernel 4: output projection out = x*Wo^T + bo ----------------
// ctx viewed flat as [NROWS, 1024] (faithful torch-.view semantics: free, it is
// contiguous). grid = (NROWS/128, 1024/64), block = 256 (8 waves).
__global__ void oproj_kernel(const bf16* __restrict__ Xc, const bf16* __restrict__ Wo,
                             const float* __restrict__ bo, float* __restrict__ out) {
    const int tid  = threadIdx.x;
    const int wave = tid >> 5;
    const int lane = tid & 31;
    const int half = lane >> 4;
    const int nl   = lane & 15;
    const int r0   = blockIdx.x * 128 + wave * 16;
    const int j0   = blockIdx.y * 64;

    v8f acc[4];
#pragma unroll
    for (int j = 0; j < 4; ++j) acc[j] = vzero8();

    const bf16* arow = Xc + (size_t)(r0 + nl) * D_MODEL;

    for (int kk = 0; kk < D_MODEL; kk += 32) {
        v16bf a = load_frag(arow + kk + half * 8, arow + kk + 16 + half * 8);
#pragma unroll
        for (int j = 0; j < 4; ++j) {
            // B[k][n] = Wo[(j0+n)*D_MODEL + kk + k]  (out = x @ Wo^T)
            const bf16* bp = Wo + (size_t)(j0 + j * 16 + nl) * D_MODEL + kk + half * 8;
            v16bf b = load_frag(bp, bp + 16);
            acc[j] = wmma_bf16(a, b, acc[j]);
        }
    }

#pragma unroll
    for (int j = 0; j < 4; ++j) {
        const float bb = bo[j0 + j * 16 + nl];
#pragma unroll
        for (int r = 0; r < 8; ++r)
            out[(size_t)(r0 + r + 8 * half) * D_MODEL + j0 + j * 16 + nl] =
                acc[j][r] + bb;
    }
}

// ---------------- workspace layout (bytes) ----------------
#define WS_QBF   ((size_t)0)
#define WS_KBF   (WS_QBF + (size_t)NROWS * D_MODEL * 2)
#define WS_VBF   (WS_KBF + (size_t)NROWS * D_MODEL * 2)
#define WS_WQBF  (WS_VBF + (size_t)NROWS * D_MODEL * 2)
#define WS_WKBF  (WS_WQBF + (size_t)NUM_HEADS * D_MODEL * D_K * 2)
#define WS_WVBF  (WS_WKBF + (size_t)NUM_HEADS * D_MODEL * D_K * 2)
#define WS_WOBF  (WS_WVBF + (size_t)NUM_HEADS * D_MODEL * D_K * 2)
#define WS_QH    (WS_WOBF + (size_t)D_MODEL * D_MODEL * 2)
#define WS_KH    (WS_QH + (size_t)NROWS * D_MODEL * 2)
#define WS_VH    (WS_KH + (size_t)NROWS * D_MODEL * 2)
#define WS_CTX   (WS_VH + (size_t)NROWS * D_MODEL * 2)

extern "C" void kernel_launch(void* const* d_in, const int* in_sizes, int n_in,
                              void* d_out, int out_size, void* d_ws, size_t ws_size,
                              hipStream_t stream) {
    const float* Q  = (const float*)d_in[0];
    const float* K  = (const float*)d_in[1];
    const float* V  = (const float*)d_in[2];
    const float* Wq = (const float*)d_in[3];
    const float* bq = (const float*)d_in[4];
    const float* Wk = (const float*)d_in[5];
    const float* bk = (const float*)d_in[6];
    const float* Wv = (const float*)d_in[7];
    const float* bv = (const float*)d_in[8];
    const float* Wo = (const float*)d_in[9];
    const float* bo = (const float*)d_in[10];
    float* out = (float*)d_out;

    char* ws = (char*)d_ws;
    bf16* Qbf  = (bf16*)(ws + WS_QBF);
    bf16* Kbf  = (bf16*)(ws + WS_KBF);
    bf16* Vbf  = (bf16*)(ws + WS_VBF);
    bf16* Wqbf = (bf16*)(ws + WS_WQBF);
    bf16* Wkbf = (bf16*)(ws + WS_WKBF);
    bf16* Wvbf = (bf16*)(ws + WS_WVBF);
    bf16* Wobf = (bf16*)(ws + WS_WOBF);
    bf16* Qh   = (bf16*)(ws + WS_QH);
    bf16* Kh   = (bf16*)(ws + WS_KH);
    bf16* Vh   = (bf16*)(ws + WS_VH);
    bf16* Ctx  = (bf16*)(ws + WS_CTX);

    const int nAct = NROWS * D_MODEL;            // 4,194,304
    const int nW   = NUM_HEADS * D_MODEL * D_K;  // 1,048,576
    const int nWo  = D_MODEL * D_MODEL;          // 1,048,576

    cvt_f32_bf16_kernel<<<4096, 256, 0, stream>>>(Q,  Qbf,  nAct);
    cvt_f32_bf16_kernel<<<4096, 256, 0, stream>>>(K,  Kbf,  nAct);
    cvt_f32_bf16_kernel<<<4096, 256, 0, stream>>>(V,  Vbf,  nAct);
    cvt_f32_bf16_kernel<<<2048, 256, 0, stream>>>(Wq, Wqbf, nW);
    cvt_f32_bf16_kernel<<<2048, 256, 0, stream>>>(Wk, Wkbf, nW);
    cvt_f32_bf16_kernel<<<2048, 256, 0, stream>>>(Wv, Wvbf, nW);
    cvt_f32_bf16_kernel<<<2048, 256, 0, stream>>>(Wo, Wobf, nWo);

    proj_kernel<<<dim3(NROWS / 128, NUM_HEADS, 3), 256, 0, stream>>>(
        Qbf, Kbf, Vbf, Wqbf, Wkbf, Wvbf, bq, bk, bv, Qh, Kh, Vh);

    attn_kernel<<<BATCH * NUM_HEADS * (SEQ / 128), 256, 0, stream>>>(Qh, Kh, Vh, Ctx);

    oproj_kernel<<<dim3(NROWS / 128, D_MODEL / 64), 256, 0, stream>>>(
        Ctx, Wobf, bo, out);
}
